// SingleHeadAttention_13606456394155
// MI455X (gfx1250) — matrix operand order repaired and checked by reading, not compiled, not run on hardware
//
#include <hip/hip_runtime.h>

typedef __attribute__((ext_vector_type(16))) __bf16 v16bf;
typedef __attribute__((ext_vector_type(8)))  float  v8f;

#define BB 4
#define SS 4096
#define DD 256
#define SCALE_Q 0.0625f   // 1/sqrt(256)

// 16x32 bf16 A-fragment from an fp32 row (contiguous K, k = k0 + 16*hi + s: lane&15 = row,
// VGPR pairs hold contiguous K pairs, lane-half selects K group).
__device__ __forceinline__ v16bf a_frag_f32(const float* arow, int k0, int hi)
{
    v16bf a;
    #pragma unroll
    for (int hv = 0; hv < 8; ++hv) {
        int kk = k0 + 2 * hv + hi * 16;
        float2 f2 = *(const float2*)(arow + kk);
        a[2 * hv]     = (__bf16)f2.x;
        a[2 * hv + 1] = (__bf16)f2.y;
    }
    return a;
}

// Same A-fragment from an already-bf16 row.
__device__ __forceinline__ v16bf a_frag_bf16(const __bf16* arow, int k0, int hi)
{
    v16bf a;
    #pragma unroll
    for (int hv = 0; hv < 8; ++hv) {
        int kk = k0 + 2 * hv + hi * 16;
        a[2 * hv]     = arow[kk];
        a[2 * hv + 1] = arow[kk + 1];
    }
    return a;
}

// ---------------------------------------------------------------------------
// fp32 -> bf16 bulk convert (X and the four weight matrices, done once).
// ---------------------------------------------------------------------------
__global__ __launch_bounds__(256)
void cvt_bf16_kernel(const float* __restrict__ src, __bf16* __restrict__ dst, int n)
{
    const int stride = gridDim.x * 256;
    const float2* s2 = (const float2*)src;
    for (int i = blockIdx.x * 256 + threadIdx.x; i < n / 2; i += stride) {
        float2 f = s2[i];
        dst[2 * i]     = (__bf16)f.x;
        dst[2 * i + 1] = (__bf16)f.y;
    }
}

// ---------------------------------------------------------------------------
// Stage 1: Q/K/V projection, all-bf16 operands.  16x64 tile per wave.
// B-fragments preloaded into an array so the 4 WMMAs issue back-to-back
// (one WAR-hazard gap per group instead of per WMMA).
// V stored transposed: Vt[b][d][s].
// ---------------------------------------------------------------------------
__global__ __launch_bounds__(32)
void qkv_proj_kernel(const __bf16* __restrict__ Xb,
                     const __bf16* __restrict__ Wqb, const float* __restrict__ bq,
                     const __bf16* __restrict__ Wkb, const float* __restrict__ bk,
                     const __bf16* __restrict__ Wvb, const float* __restrict__ bv,
                     __bf16* __restrict__ Qb, __bf16* __restrict__ Kb,
                     __bf16* __restrict__ Vt)
{
    const int lane = threadIdx.x;
    const int row0 = blockIdx.x * 16;   // row in flattened [B*S]
    const int col0 = blockIdx.y * 64;
    const __bf16* W; const float* bias; float oscale;
    if (blockIdx.z == 0)      { W = Wqb; bias = bq; oscale = SCALE_Q; }
    else if (blockIdx.z == 1) { W = Wkb; bias = bk; oscale = 1.0f; }
    else                      { W = Wvb; bias = bv; oscale = 1.0f; }

    const int n  = lane & 15;
    const int hi = lane >> 4;
    const __bf16* arow = Xb + (size_t)(row0 + n) * DD;

    v8f acc[4] = {};
    #pragma unroll
    for (int k0 = 0; k0 < DD; k0 += 32) {
        v16bf a = a_frag_bf16(arow, k0, hi);
        v16bf bfr[4];
        #pragma unroll
        for (int j = 0; j < 4; ++j)
            bfr[j] = *(const v16bf*)(W + (size_t)(col0 + 16 * j + n) * DD +
                                     hi * 16 + k0);
        #pragma unroll
        for (int j = 0; j < 4; ++j)
            acc[j] = __builtin_amdgcn_wmma_f32_16x16x32_bf16(
                false, a, false, bfr[j], (short)0, acc[j], false, false);
    }

    const int bidx = row0 / SS;
    const int s0   = (row0 % SS) + hi * 8;
    #pragma unroll
    for (int j = 0; j < 4; ++j) {
        const int col = col0 + 16 * j + n;
        const float bb = bias[col];
        #pragma unroll
        for (int r = 0; r < 8; ++r) {
            float v = (acc[j][r] + bb) * oscale;
            if (blockIdx.z == 0)
                Qb[(size_t)(row0 + hi * 8 + r) * DD + col] = (__bf16)v;
            else if (blockIdx.z == 1)
                Kb[(size_t)(row0 + hi * 8 + r) * DD + col] = (__bf16)v;
            else
                Vt[(size_t)bidx * DD * SS + (size_t)col * SS + s0 + r] = (__bf16)v;
        }
    }
}

// ---------------------------------------------------------------------------
// Fused attention core: scores + mask + softmax + att@V for one 16-query
// tile, per workgroup (16 waves / 512 threads, 256KB LDS att panel).
//   Phase A: wave w -> scores[16 x 256] for keys [256w,256w+256) into LDS
//   Phase B: wave w -> softmax of LDS row w, normalized row written ONCE
//            (coalesced) to the att output
//   Phase C: wave w -> out tile [16 x 16] for d-cols [16w,16w+16) from the
//            LDS panel x Vt (contiguous bf16 rows)
// ---------------------------------------------------------------------------
__global__ __launch_bounds__(512)
void attention_fused_kernel(const __bf16* __restrict__ Qb,
                            const __bf16* __restrict__ Kb,
                            const __bf16* __restrict__ Vt,
                            const int* __restrict__ mask,
                            float* __restrict__ att,
                            __bf16* __restrict__ Ob)
{
    extern __shared__ float satt[];            // [16][SS] fp32 = 256 KB
    const int tid  = threadIdx.x;
    const int w    = tid >> 5;                 // wave id 0..15
    const int ln   = tid & 31;                 // lane id
    const int n    = ln & 15;
    const int hi   = ln >> 4;
    const int q0   = blockIdx.x * 16;
    const int b    = blockIdx.y;

    const __bf16* Q = Qb + (size_t)b * SS * DD;
    const __bf16* K = Kb + (size_t)b * SS * DD;

    // ---- Phase A: scores into LDS -------------------------------------
    {
        const __bf16* arow = Q + (size_t)(q0 + n) * DD;
        v16bf aQ[8];
        #pragma unroll
        for (int kc = 0; kc < 8; ++kc)         // cache whole Q tile: 64 VGPRs
            aQ[kc] = a_frag_bf16(arow, kc * 32, hi);

        for (int kt = 0; kt < 16; ++kt) {      // 16 key sub-tiles of 16
            const int t0 = w * 256 + kt * 16;
            const __bf16* brow = K + (size_t)(t0 + n) * DD + hi * 16;
            v8f acc = {};
            #pragma unroll
            for (int kc = 0; kc < 8; ++kc) {
                v16bf bfr = *(const v16bf*)(brow + kc * 32);
                acc = __builtin_amdgcn_wmma_f32_16x16x32_bf16(
                    false, aQ[kc], false, bfr, (short)0, acc, false, false);
            }
            const int kcol = t0 + n;
            const int mk = mask[b * SS + kcol];
            #pragma unroll
            for (int r = 0; r < 8; ++r) {
                float s = (mk == 0) ? -1e20f : acc[r];
                satt[(hi * 8 + r) * SS + kcol] = s;
            }
        }
    }
    __syncthreads();

    // ---- Phase B: softmax of row w, write att output once -------------
    {
        float* row = satt + w * SS;
        float mx = -3.4e38f;
        for (int i = ln; i < SS; i += 32) mx = fmaxf(mx, row[i]);
        #pragma unroll
        for (int off = 16; off > 0; off >>= 1)
            mx = fmaxf(mx, __shfl_xor(mx, off, 32));

        float sum = 0.f;
        for (int i = ln; i < SS; i += 32) {
            float e = __expf(row[i] - mx);
            row[i] = e;
            sum += e;
        }
        #pragma unroll
        for (int off = 16; off > 0; off >>= 1)
            sum += __shfl_xor(sum, off, 32);

        const float inv = 1.0f / sum;
        float* gatt = att + (size_t)b * SS * SS + (size_t)(q0 + w) * SS;
        for (int i = ln; i < SS; i += 32) {
            float v = row[i] * inv;
            row[i] = v;                        // normalized panel for phase C
            gatt[i] = v;                       // coalesced output write
        }
    }
    __syncthreads();

    // ---- Phase C: out[16 x 16] for d-cols [16w,16w+16) ----------------
    {
        const int d0 = w * 16;
        const float*  arow = satt + (size_t)n * SS;              // LDS row
        const __bf16* vrow = Vt + (size_t)b * DD * SS +
                             (size_t)(d0 + n) * SS + hi * 16;
        v8f acc = {};
        for (int k0 = 0; k0 < SS; k0 += 32) {
            __builtin_prefetch(vrow + k0 + 128, 0, 0);           // global_prefetch
            v16bf a = a_frag_f32(arow, k0, hi);                  // ds_load + cvt
            v16bf bfr = *(const v16bf*)(vrow + k0);
            acc = __builtin_amdgcn_wmma_f32_16x16x32_bf16(
                false, a, false, bfr, (short)0, acc, false, false);
        }
        #pragma unroll
        for (int r = 0; r < 8; ++r)
            Ob[(size_t)b * SS * DD + (size_t)(q0 + hi * 8 + r) * DD + d0 + n] =
                (__bf16)acc[r];
    }
}

// ---------------------------------------------------------------------------
// Stage 3: ret = out @ Wo^T + bo  (fp32 output), all-bf16 operands.
// ---------------------------------------------------------------------------
__global__ __launch_bounds__(32)
void out_proj_kernel(const __bf16* __restrict__ Ob, const __bf16* __restrict__ Wob,
                     const float* __restrict__ bo, float* __restrict__ ret)
{
    const int lane = threadIdx.x;
    const int row0 = blockIdx.x * 16;
    const int col0 = blockIdx.y * 64;
    const int n  = lane & 15;
    const int hi = lane >> 4;
    const __bf16* arow = Ob + (size_t)(row0 + n) * DD;

    v8f acc[4] = {};
    #pragma unroll
    for (int k0 = 0; k0 < DD; k0 += 32) {
        v16bf a = a_frag_bf16(arow, k0, hi);
        v16bf bfr[4];
        #pragma unroll
        for (int j = 0; j < 4; ++j)
            bfr[j] = *(const v16bf*)(Wob + (size_t)(col0 + 16 * j + n) * DD +
                                     hi * 16 + k0);
        #pragma unroll
        for (int j = 0; j < 4; ++j)
            acc[j] = __builtin_amdgcn_wmma_f32_16x16x32_bf16(
                false, a, false, bfr[j], (short)0, acc[j], false, false);
    }
    #pragma unroll
    for (int j = 0; j < 4; ++j) {
        const int col = col0 + 16 * j + n;
        const float bb = bo[col];
        #pragma unroll
        for (int r = 0; r < 8; ++r)
            ret[(size_t)(row0 + hi * 8 + r) * DD + col] = acc[j][r] + bb;
    }
}

// ---------------------------------------------------------------------------
extern "C" void kernel_launch(void* const* d_in, const int* in_sizes, int n_in,
                              void* d_out, int out_size, void* d_ws, size_t ws_size,
                              hipStream_t stream)
{
    const float* X    = (const float*)d_in[0];
    const int*   mask = (const int*)d_in[1];
    const float* Wq   = (const float*)d_in[2];
    const float* bq   = (const float*)d_in[3];
    const float* Wk   = (const float*)d_in[4];
    const float* bk   = (const float*)d_in[5];
    const float* Wv   = (const float*)d_in[6];
    const float* bv   = (const float*)d_in[7];
    const float* Wo   = (const float*)d_in[8];
    const float* bo   = (const float*)d_in[9];

    float* ret = (float*)d_out;                         // [B,S,D]
    float* att = ret + (size_t)BB * SS * DD;            // [B,S,S]

    const size_t elems = (size_t)BB * SS * DD;          // 4,194,304
    __bf16* Qb  = (__bf16*)d_ws;                        // 8 MB each
    __bf16* Kb  = Qb + elems;
    __bf16* Vt  = Kb + elems;                           // transposed [B,D,S]
    __bf16* Ob  = Vt + elems;
    __bf16* Xb  = Ob + elems;
    __bf16* Wqb = Xb + elems;                           // 128 KB each
    __bf16* Wkb = Wqb + DD * DD;
    __bf16* Wvb = Wkb + DD * DD;
    __bf16* Wob = Wvb + DD * DD;                        // ~40.5 MB total

    cvt_bf16_kernel<<<1024, 256, 0, stream>>>(X,  Xb,  (int)elems);
    cvt_bf16_kernel<<<64,   256, 0, stream>>>(Wq, Wqb, DD * DD);
    cvt_bf16_kernel<<<64,   256, 0, stream>>>(Wk, Wkb, DD * DD);
    cvt_bf16_kernel<<<64,   256, 0, stream>>>(Wv, Wvb, DD * DD);
    cvt_bf16_kernel<<<64,   256, 0, stream>>>(Wo, Wob, DD * DD);

    qkv_proj_kernel<<<dim3(BB * SS / 16, DD / 64, 3), 32, 0, stream>>>(
        Xb, Wqb, bq, Wkb, bk, Wvb, bv, Qb, Kb, Vt);

    attention_fused_kernel<<<dim3(SS / 16, BB), 512, 16 * SS * sizeof(float),
                             stream>>>(Qb, Kb, Vt, mask, att, Ob);

    out_proj_kernel<<<dim3(BB * SS / 16, DD / 64), 32, 0, stream>>>(
        Ob, Wob, bo, ret);
}
